// ConsecutiveIntentUnit_13915694039214
// MI455X (gfx1250) — compile-verified, matrix operand order checked by reading, not run on hardware
//
#include <hip/hip_runtime.h>

#define DEVI __device__ __forceinline__
#define AS1 __attribute__((address_space(1)))
#define AS3 __attribute__((address_space(3)))

typedef __attribute__((ext_vector_type(16))) __bf16 v16bf;
typedef __attribute__((ext_vector_type(8)))  float  v8f;
typedef int v4i_vs __attribute__((vector_size(16))); // matches builtin's pointee type

constexpr int B = 1024, S = 200, E = 128, H = 256, G = 384; // G = 3*E

// K index inside a 16x32 bf16 A/B fragment for (lane, element e in v16bf).
// Per ISA 7.12.2: lanes 0-15 hold K {0..7,16..23}, lanes 16-31 hold K {8..15,24..31}.
DEVI int wmma_k(int lane, int e) {
  int j = e >> 1;
  return ((j & 4) << 2) | ((lane >> 4) << 3) | ((j & 3) << 1) | (e & 1);
}

DEVI v8f wmma_bf16(v16bf a, v16bf b, v8f c) {
  return __builtin_amdgcn_wmma_f32_16x16x32_bf16(false, a, false, b, (short)0, c,
                                                 false, false);
}

DEVI float sigmoidf_(float x) { return 1.0f / (1.0f + __expf(-x)); }

// ---- CDNA5 async global->LDS copy (ASYNCcnt-tracked), with safe fallback ----
DEVI void async_copy_b128(const void* g, void* l) {
#if __has_builtin(__builtin_amdgcn_global_load_async_to_lds_b128)
  void* gg = const_cast<void*>(g);
  __builtin_amdgcn_global_load_async_to_lds_b128((AS1 v4i_vs*)gg, (AS3 v4i_vs*)l,
                                                 0, 0);
#else
  *(float4*)l = *(const float4*)g;
#endif
}

template <int N>
DEVI void wait_async() {
#if __has_builtin(__builtin_amdgcn_s_wait_asynccnt)
  __builtin_amdgcn_s_wait_asynccnt((unsigned short)N);
#elif __has_builtin(__builtin_amdgcn_global_load_async_to_lds_b128)
  asm volatile("s_wait_asynccnt %0" ::"i"(N));
#endif
}

// ---------------------------------------------------------------------------
// Pack a row-major fp32 weight W[N][Kt] into bf16 WMMA B-fragments:
// frag = ntile*Ktiles + ktile, 512 bf16 per frag, lane-contiguous (32B/lane).
// For C = A @ W^T, B[k][n] = W[n][k].
// ---------------------------------------------------------------------------
__global__ void pack_wfrags(const float* __restrict__ W, __bf16* __restrict__ out,
                            int Kt, int Ktiles) {
  int frag  = blockIdx.x;
  int ntile = frag / Ktiles;
  int ktile = frag - ntile * Ktiles;
  int lane  = threadIdx.x;
  int n = ntile * 16 + (lane & 15);
  v16bf v;
#pragma unroll
  for (int e = 0; e < 16; ++e)
    v[e] = (__bf16)W[(size_t)n * Kt + ktile * 32 + wmma_k(lane, e)];
  *(v16bf*)(out + (size_t)frag * 512 + lane * 16) = v;
}

// ---------------------------------------------------------------------------
// pooled[b][e] = mean_s x[b][s][e]
// ---------------------------------------------------------------------------
__global__ void pool_mean(const float* __restrict__ x, float* __restrict__ pooled) {
  int idx = blockIdx.x * blockDim.x + threadIdx.x; // b*128 + e
  int b = idx >> 7, e = idx & 127;
  const float* p = x + (size_t)b * S * E + e;
  float acc = 0.f;
  for (int s = 0; s < S; ++s) acc += p[(size_t)s * E];
  pooled[idx] = acc * (1.0f / (float)S);
}

// ---------------------------------------------------------------------------
// gi[s][b][g] = x[b][s][:] @ W_ih^T + b_ih  (20 GFLOP GEMM, x read once)
// ---------------------------------------------------------------------------
__global__ void gi_gemm(const float* __restrict__ x, const __bf16* __restrict__ Wf,
                        const float* __restrict__ bih, float* __restrict__ gi) {
  int bt = blockIdx.x;            // 0..63
  int s  = blockIdx.y;            // 0..199
  int lane = threadIdx.x;
  int nlo  = lane & 15;
  int mloc = (lane >> 4) << 3;

  const float* arow = x + ((size_t)(bt * 16 + nlo) * S + s) * E;
  v16bf a[4];
#pragma unroll
  for (int kt = 0; kt < 4; ++kt)
#pragma unroll
    for (int e = 0; e < 16; ++e)
      a[kt][e] = (__bf16)arow[kt * 32 + wmma_k(lane, e)];

  for (int nt = 0; nt < 24; ++nt) {
    v8f acc = {};
#pragma unroll
    for (int kt = 0; kt < 4; ++kt) {
      v16bf bfr = *(const v16bf*)(Wf + (size_t)(nt * 4 + kt) * 512 + lane * 16);
      acc = wmma_bf16(a[kt], bfr, acc);
    }
    int col = nt * 16 + nlo;
    float bias = bih[col];
#pragma unroll
    for (int r = 0; r < 8; ++r) {
      int m = bt * 16 + mloc + r;
      gi[((size_t)s * B + m) * G + col] = acc[r] + bias;
    }
  }
}

// ---------------------------------------------------------------------------
// Generic small GEMM: out[:, colOff:] = act(A[M][K] @ W^T + bias)
// ---------------------------------------------------------------------------
template <int ACT, int KT, int NT>
__global__ void gemm_wmma_kernel(const float* __restrict__ A,
                                 const __bf16* __restrict__ Wf,
                                 const float* __restrict__ bias,
                                 float* __restrict__ out,
                                 int outStride, int colOffset) {
  int mt = blockIdx.x;
  int lane = threadIdx.x;
  int nlo  = lane & 15;
  int mloc = (lane >> 4) << 3;
  const int Kdim = KT * 32;

  const float* arow = A + (size_t)(mt * 16 + nlo) * Kdim;
  v16bf a[KT];
#pragma unroll
  for (int kt = 0; kt < KT; ++kt)
#pragma unroll
    for (int e = 0; e < 16; ++e)
      a[kt][e] = (__bf16)arow[kt * 32 + wmma_k(lane, e)];

  for (int nt = 0; nt < NT; ++nt) {
    v8f acc = {};
#pragma unroll
    for (int kt = 0; kt < KT; ++kt) {
      v16bf bfr = *(const v16bf*)(Wf + (size_t)(nt * KT + kt) * 512 + lane * 16);
      acc = wmma_bf16(a[kt], bfr, acc);
    }
    int col = nt * 16 + nlo;
    float bb = bias[col];
#pragma unroll
    for (int r = 0; r < 8; ++r) {
      float v = acc[r] + bb;
      if (ACT == 1) v = fmaxf(v, 0.0f);
      else if (ACT == 2) v = tanhf(v);
      out[(size_t)(mt * 16 + mloc + r) * outStride + colOffset + col] = v;
    }
  }
}

// ---------------------------------------------------------------------------
// GRU scan, 4 waves / 16 batch rows:
//   waves 0..2: one gate GEMM each (8 ntiles x 4 ktiles = 32 WMMAs/step)
//   wave 3:     async-DMA double-buffer staging of next step's gi slice
// LDS (180 KB, CDNA5 320KB/WGP): W_hh frags 96K | h bf16 4K | h f32 8K |
//                                gh 24K | gi double buffer 48K
// ---------------------------------------------------------------------------
__global__ void gru_scan(const float* __restrict__ gi,
                         const __bf16* __restrict__ whhFragsG,
                         const float* __restrict__ bhh,
                         float* __restrict__ cat) {
  extern __shared__ char smem[];
  __bf16* whh   = (__bf16*)smem;                 // 96 KB, 96 frags
  __bf16* hbuf  = (__bf16*)(smem + 98304);       // 16x128 bf16
  float*  hf32  = (float*)(smem + 102400);       // 16x128 f32
  float*  ghbuf = (float*)(smem + 110592);       // 3 x 16x128 f32
  float*  gibuf = (float*)(smem + 135168);       // 2 x 16x384 f32

  int tid  = threadIdx.x;
  int lane = tid & 31;
  int w    = tid >> 5;
  int b0   = blockIdx.x * 16;
  int nlo  = lane & 15;
  int mloc = (lane >> 4) << 3;

  // ---- preamble: async-stage W_hh frags (96KB) + gi step-0 slice (24KB) ----
  for (int i = tid; i < 98304 / 16; i += 128)
    async_copy_b128((const char*)whhFragsG + i * 16, (char*)whh + i * 16);
  {
    const float* g0 = gi + (size_t)b0 * G; // s = 0 slice: contiguous 16x384 f32
    for (int i = tid; i < 24576 / 16; i += 128)
      async_copy_b128((const char*)g0 + i * 16, (char*)gibuf + i * 16);
  }
  for (int i = tid; i < 16 * 128; i += 128) {
    hf32[i] = 0.0f;
    hbuf[i] = (__bf16)0.0f;
  }
  wait_async<0>();
  __syncthreads();

  // Per-thread column (tid = col 0..127) biases, hoisted out of the scan.
  float bhr = bhh[tid];
  float bhz = bhh[128 + tid];
  float bhn = bhh[256 + tid];

  for (int s = 0; s < S; ++s) {
    // ---------------- Phase A ----------------
    if (w < 3) {
      // gather h into 4 A-fragments from LDS
      v16bf ah[4];
#pragma unroll
      for (int kt = 0; kt < 4; ++kt)
#pragma unroll
        for (int e = 0; e < 16; ++e)
          ah[kt][e] = hbuf[nlo * 128 + kt * 32 + wmma_k(lane, e)];

#pragma unroll
      for (int j = 0; j < 8; ++j) {
        v8f acc = {};
#pragma unroll
        for (int kt = 0; kt < 4; ++kt) {
          v16bf bfr =
              *(const v16bf*)(whh + (size_t)((w * 8 + j) * 4 + kt) * 512 + lane * 16);
          acc = wmma_bf16(ah[kt], bfr, acc);
        }
#pragma unroll
        for (int r = 0; r < 8; ++r)
          ghbuf[w * 2048 + (mloc + r) * 128 + j * 16 + nlo] = acc[r];
      }
    } else {
      // DMA wave: stage next step's gi slice, then counted wait so the
      // CURRENT buffer (issued last step) is complete before the barrier.
      if (s + 1 < S) {
        const float* gsrc = gi + ((size_t)(s + 1) * B + b0) * G;
        float* ldst = gibuf + ((s + 1) & 1) * 6144;
        for (int i = lane; i < 1536; i += 32)
          async_copy_b128((const char*)gsrc + i * 16, (char*)ldst + i * 16);
        wait_async<48>(); // 48 just issued; older 48 (current buffer) retired
      } else {
        wait_async<0>();
      }
    }
    __syncthreads();

    // ---------------- Phase B: elementwise gate math, col = tid ----------------
    const float* gib = gibuf + (s & 1) * 6144;
#pragma unroll
    for (int m = 0; m < 16; ++m) {
      float ghr = ghbuf[m * 128 + tid];
      float ghz = ghbuf[2048 + m * 128 + tid];
      float ghn = ghbuf[4096 + m * 128 + tid];
      float gir = gib[m * 384 + tid];
      float giz = gib[m * 384 + 128 + tid];
      float gin = gib[m * 384 + 256 + tid];
      float rg = sigmoidf_(gir + ghr + bhr);
      float zg = sigmoidf_(giz + ghz + bhz);
      float ng = tanhf(gin + rg * (ghn + bhn));
      float hn = (1.0f - zg) * ng + zg * hf32[m * 128 + tid];
      hf32[m * 128 + tid] = hn;
      hbuf[m * 128 + tid] = (__bf16)hn;
    }
    __syncthreads();
  }

  // seq_repr -> cat[:, 128:256]
#pragma unroll
  for (int m = 0; m < 16; ++m)
    cat[(size_t)(b0 + m) * 256 + 128 + tid] = hf32[m * 128 + tid];
}

// ---------------------------------------------------------------------------
extern "C" void kernel_launch(void* const* d_in, const int* in_sizes, int n_in,
                              void* d_out, int out_size, void* d_ws, size_t ws_size,
                              hipStream_t stream) {
  const float* x   = (const float*)d_in[0];
  const float* W1  = (const float*)d_in[1];
  const float* b1  = (const float*)d_in[2];
  const float* W2  = (const float*)d_in[3];
  const float* b2  = (const float*)d_in[4];
  const float* Wih = (const float*)d_in[5];
  const float* bih = (const float*)d_in[6];
  const float* Whh = (const float*)d_in[7];
  const float* bhh = (const float*)d_in[8];
  const float* Wc  = (const float*)d_in[9];
  const float* bc  = (const float*)d_in[10];
  float* out = (float*)d_out;

  char* ws = (char*)d_ws;
  size_t off = 0;
  float*  gi     = (float*)(ws + off); off += (size_t)S * B * G * 4; // 300 MB
  float*  pooled = (float*)(ws + off); off += (size_t)B * E * 4;
  float*  tbuf   = (float*)(ws + off); off += (size_t)B * H * 4;
  float*  cat    = (float*)(ws + off); off += (size_t)B * 2 * E * 4;
  __bf16* fWih   = (__bf16*)(ws + off); off += 96 * 512 * 2;
  __bf16* fWhh   = (__bf16*)(ws + off); off += 96 * 512 * 2;
  __bf16* fW1    = (__bf16*)(ws + off); off += 64 * 512 * 2;
  __bf16* fW2    = (__bf16*)(ws + off); off += 64 * 512 * 2;
  __bf16* fWc    = (__bf16*)(ws + off); off += 64 * 512 * 2;

  // Prepack weights into WMMA fragment order (bf16).
  pack_wfrags<<<96, 32, 0, stream>>>(Wih, fWih, 128, 4); // N=384,K=128
  pack_wfrags<<<96, 32, 0, stream>>>(Whh, fWhh, 128, 4); // N=384,K=128
  pack_wfrags<<<64, 32, 0, stream>>>(W1,  fW1,  128, 4); // N=256,K=128
  pack_wfrags<<<64, 32, 0, stream>>>(W2,  fW2,  256, 8); // N=128,K=256
  pack_wfrags<<<64, 32, 0, stream>>>(Wc,  fWc,  256, 8); // N=128,K=256

  // Set path + input projection.
  pool_mean<<<(B * E) / 256, 256, 0, stream>>>(x, pooled);
  gi_gemm<<<dim3(B / 16, S), 32, 0, stream>>>(x, fWih, bih, gi);
  gemm_wmma_kernel<1, 4, 16><<<B / 16, 32, 0, stream>>>(pooled, fW1, b1, tbuf, H, 0);
  gemm_wmma_kernel<0, 8, 8><<<B / 16, 32, 0, stream>>>(tbuf, fW2, b2, cat, 2 * E, 0);

  // Recurrence (fills cat[:,128:]), then final combine.
  gru_scan<<<B / 16, 128, 184320, stream>>>(gi, fWhh, bhh, cat);
  gemm_wmma_kernel<2, 8, 8><<<B / 16, 32, 0, stream>>>(cat, fWc, bc, out, E, 0);
}
